// Model_83141976916118
// MI455X (gfx1250) — compile-verified, matrix operand order checked by reading, not run on hardware
//
#include <hip/hip_runtime.h>

typedef __attribute__((ext_vector_type(16))) _Float16 v16h;
typedef __attribute__((ext_vector_type(8)))  _Float16 v8h;
typedef __attribute__((ext_vector_type(8)))  float    v8f;

#define T_  1024
#define B_  64
#define I_  128
#define H_  256
#define L_  6
#define G_  768            // 3H
#define TB_ 65536          // T*B

// Element e (0..15) of a 16x32 f16 A/B fragment held by a lane with sub = lane>>4:
// k = (v<4 ? 2v : 16+2(v-4)) + (e&1) + 8*sub, v = e>>1  (per CDNA5 ISA 7.12.2)
// => per lane the 16 k's are two contiguous runs: [8s,8s+8) and [16+8s,16+8s+8)
__device__ __forceinline__ int kmap(int e, int s) {
  int v = e >> 1;
  return ((v < 4) ? (2 * v) : (16 + 2 * (v - 4))) + (e & 1) + 8 * s;
}

// ---------------------------------------------------------------- cast f32->f16
__global__ void cast_f32_f16(const float* __restrict__ src,
                             _Float16* __restrict__ dst, int n) {
  int i = blockIdx.x * blockDim.x + threadIdx.x;
  if (i < n) dst[i] = (_Float16)src[i];
}

// ------------------------------------------------- pre-swizzle Whh to B-fragment
// layout: [l*2+d][ct(48)][kc(8)][lane(32)][e(16)] f16 -> each lane reads a whole
// 16-half B fragment as two contiguous b128 loads every timestep (L2-resident).
__global__ void swizzle_whh(const float* __restrict__ whh,   // [12][768][256]
                            _Float16* __restrict__ frag) {
  int i = blockIdx.x * blockDim.x + threadIdx.x;   // 12*48*8*32*16 = 2359296
  if (i >= 12 * 48 * 8 * 32 * 16) return;
  int e    = i & 15;
  int lane = (i >> 4) & 31;
  int kc   = (i >> 9) & 7;
  int ct   = (i >> 12) % 48;
  int ld   = i / (48 * 8 * 32 * 16);
  int n = ct * 16 + (lane & 15);
  int k = kc * 32 + kmap(e, lane >> 4);
  frag[i] = (_Float16)whh[((size_t)ld * G_ + n) * H_ + k];
}

// ---------------------------------------------------------------- xg GEMM (WMMA)
// y[M=65536, N=1536] = A[M,K] * Wc[N,K]^T + bih ; stored f16 as xg[d][m][g]
// block tile 128x64, 8 waves: wave w -> rows w*16..+16, 4 col tiles (64 cols)
__global__ __launch_bounds__(256) void gemm_xg(
    const _Float16* __restrict__ A,    // [M][K]
    const _Float16* __restrict__ Wc,   // [1536][K]
    int K,
    const float* __restrict__ bihL,    // [2*768] f32
    _Float16* __restrict__ xg)         // [2][TB][768]
{
  __shared__ _Float16 sA[128 * 32];    // 8 KB
  __shared__ _Float16 sB[64 * 32];     // 4 KB
  __shared__ _Float16 sO[128 * 64];    // 16 KB epilogue staging
  const int tid  = threadIdx.x;
  const int lane = tid & 31;
  const int w    = tid >> 5;           // 0..7
  const int ln16 = lane & 15;
  const int sub  = lane >> 4;
  const int mBase = blockIdx.x * 128;
  const int nBase = blockIdx.y * 64;   // never crosses the 768 boundary
  const int dd = nBase / G_;           // uniform per block
  const int g0 = nBase - dd * G_;

  v8f acc[4];
#pragma unroll
  for (int c = 0; c < 4; ++c)
#pragma unroll
    for (int r = 0; r < 8; ++r) acc[c][r] = 0.f;

  for (int k0 = 0; k0 < K; k0 += 32) {
    // stage A tile (128x32) : 512 x b128, 2 per thread
#pragma unroll
    for (int it = 0; it < 2; ++it) {
      int idx = tid + it * 256;
      int row = idx >> 2, seg = idx & 3;
      *(uint4*)&sA[row * 32 + seg * 8] =
          *(const uint4*)&A[(size_t)(mBase + row) * K + k0 + seg * 8];
    }
    // stage B tile (64x32) : 256 x b128, 1 per thread
    {
      int row = tid >> 2, seg = tid & 3;
      *(uint4*)&sB[row * 32 + seg * 8] =
          *(const uint4*)&Wc[(size_t)(nBase + row) * K + k0 + seg * 8];
    }
    __syncthreads();

    v16h aF;
    {
      int row = w * 16 + ln16;
#pragma unroll
      for (int p = 0; p < 8; ++p) {
        int k = ((p < 4) ? 2 * p : 16 + 2 * (p - 4)) + 8 * sub;
        aF[2 * p]     = sA[row * 32 + k];
        aF[2 * p + 1] = sA[row * 32 + k + 1];
      }
    }
#pragma unroll
    for (int c = 0; c < 4; ++c) {
      v16h bF;
      int row = c * 16 + ln16;
#pragma unroll
      for (int p = 0; p < 8; ++p) {
        int k = ((p < 4) ? 2 * p : 16 + 2 * (p - 4)) + 8 * sub;
        bF[2 * p]     = sB[row * 32 + k];
        bF[2 * p + 1] = sB[row * 32 + k + 1];
      }
      acc[c] = __builtin_amdgcn_wmma_f32_16x16x32_f16(
          false, aF, false, bF, (short)0, acc[c], false, false);
    }
    __syncthreads();
  }

  // epilogue: bias add, restage through LDS, coalesced b128 row stores
#pragma unroll
  for (int c = 0; c < 4; ++c) {
    int ncol = c * 16 + ln16;
    float bias = bihL[dd * G_ + g0 + ncol];
#pragma unroll
    for (int r = 0; r < 8; ++r) {
      int mrow = w * 16 + r + 8 * sub;
      sO[mrow * 64 + ncol] = (_Float16)(acc[c][r] + bias);
    }
  }
  __syncthreads();
  {
    _Float16* outBase = xg + (size_t)dd * TB_ * G_ + (size_t)mBase * G_ + g0;
#pragma unroll
    for (int it = 0; it < 4; ++it) {
      int idx = tid + it * 256;       // 0..1023
      int row = idx >> 2, seg = idx & 3;
      *(uint4*)&outBase[(size_t)row * G_ + seg * 8] =
          *(const uint4*)&sO[row * 64 + seg * 8];
    }
  }
}

// --------------------------------------------------------------- GRU recurrence
// grid = 2 blocks (dir), block = 1024 threads = 32 waves.
// h kept in LDS in A-fragment layout [rw(4)][kc(8)][lane(32)][e(16)] f16.
// per step: gh = h @ Whh^T + bhh (48 WMMA/wave, bias folded at store),
// barrier, fully vectorized fused gate phase updates h and streams output.
__global__ __launch_bounds__(1024) void gru_scan(
    const _Float16* __restrict__ xg,    // [2][TB][768]
    const _Float16* __restrict__ whhF,  // [2][48][8][32][16]
    const float* __restrict__ bhhL,     // [2][768]
    const float* __restrict__ h0L,      // [2][B][H]
    _Float16* __restrict__ outH,        // [TB][512] (non-final layers)
    float* __restrict__ outF,           // [TB][512] (final layer)
    int isFinal)
{
  __shared__ _Float16 hFrag[4 * 8 * 32 * 16];  // 32 KB
  __shared__ _Float16 gh[64 * G_];             // 96 KB
  const int d    = blockIdx.x;
  const int tid  = threadIdx.x;
  const int lane = tid & 31;
  const int w    = tid >> 5;     // 0..31
  const int ln16 = lane & 15;
  const int sub  = lane >> 4;
  const int rw   = w & 3;        // batch row tile 0..3
  const int cg   = w >> 2;       // col-tile group 0..7 (6 tiles each)

  // phase-2 / init ownership: one (rw2,kc2,lane2) fragment slot per thread
  const int lane2 = tid & 31;
  const int kc2   = (tid >> 5) & 7;
  const int rw2   = tid >> 8;
  const int b2    = rw2 * 16 + (lane2 & 15);
  const int s2    = lane2 >> 4;
  const int base2 = ((rw2 * 8 + kc2) * 32 + lane2) * 16;
  const int jb    = kc2 * 32 + 8 * s2;   // lo run base; hi run at jb+16

  const _Float16* wF   = whhF + (size_t)d * 48 * 8 * 32 * 16;
  const float*    bhhD = bhhL + d * G_;
  const _Float16* xgD  = xg + (size_t)d * TB_ * G_;

  {  // h0 init (f32 -> f16 into fragment layout)
    const float* h0d = h0L + ((size_t)d * B_ + b2) * H_;
#pragma unroll
    for (int e = 0; e < 16; ++e) {
      int j = kc2 * 32 + kmap(e, s2);
      hFrag[base2 + e] = (_Float16)h0d[j];
    }
  }
  __syncthreads();

  // per-lane time-invariant bhh for the 6 gh columns this lane stores
  float biasv[6];
#pragma unroll
  for (int j = 0; j < 6; ++j) biasv[j] = bhhD[(cg * 6 + j) * 16 + ln16];

  for (int t = 0; t < T_; ++t) {
    const int tt = d ? (T_ - 1 - t) : t;

    // prefetch next timestep's xg rows (hide HBM latency behind WMMA phase)
    {
      int tn = d ? (tt - 1) : (tt + 1);
      if (tn >= 0 && tn < T_) {
        int rowp = tid >> 4;          // 0..63
        int ch   = (tid & 15) % 12;   // 12 x 64-half chunks cover the 768 row
        __builtin_prefetch(xgD + ((size_t)tn * B_ + rowp) * G_ + ch * 64, 0, 0);
      }
    }

    // ---- phase 1: gh tiles via WMMA, bias folded at store
    v8f acc[6];
#pragma unroll
    for (int j = 0; j < 6; ++j)
#pragma unroll
      for (int r = 0; r < 8; ++r) acc[j][r] = 0.f;

    for (int kc = 0; kc < 8; ++kc) {
      v16h aF = *(const v16h*)&hFrag[((rw * 8 + kc) * 32 + lane) * 16];
#pragma unroll
      for (int j = 0; j < 6; ++j) {
        int ct = cg * 6 + j;
        v16h bF = *(const v16h*)&wF[(((size_t)ct * 8 + kc) * 32 + lane) * 16];
        acc[j] = __builtin_amdgcn_wmma_f32_16x16x32_f16(
            false, aF, false, bF, (short)0, acc[j], false, false);
      }
    }
#pragma unroll
    for (int j = 0; j < 6; ++j) {
      int n = (cg * 6 + j) * 16 + ln16;
#pragma unroll
      for (int r = 0; r < 8; ++r) {
        int b = rw * 16 + r + 8 * sub;
        gh[b * G_ + n] = (_Float16)(acc[j][r] + biasv[j]);
      }
    }
    __syncthreads();

    // ---- phase 2: fused GRU gates, all memory traffic as b128 runs
    {
      const _Float16* xgRow = xgD + ((size_t)tt * B_ + b2) * G_;
      const _Float16* ghRow = &gh[b2 * G_];
      v8h xr_lo = *(const v8h*)&xgRow[jb];        v8h xr_hi = *(const v8h*)&xgRow[jb + 16];
      v8h xz_lo = *(const v8h*)&xgRow[256 + jb];  v8h xz_hi = *(const v8h*)&xgRow[256 + jb + 16];
      v8h xn_lo = *(const v8h*)&xgRow[512 + jb];  v8h xn_hi = *(const v8h*)&xgRow[512 + jb + 16];
      v8h hr_lo = *(const v8h*)&ghRow[jb];        v8h hr_hi = *(const v8h*)&ghRow[jb + 16];
      v8h hz_lo = *(const v8h*)&ghRow[256 + jb];  v8h hz_hi = *(const v8h*)&ghRow[256 + jb + 16];
      v8h hn_lo = *(const v8h*)&ghRow[512 + jb];  v8h hn_hi = *(const v8h*)&ghRow[512 + jb + 16];
      v8h ho_lo = *(const v8h*)&hFrag[base2];     v8h ho_hi = *(const v8h*)&hFrag[base2 + 8];

      v8h nw_lo, nw_hi;
      v8f fo_lo, fo_hi;
#pragma unroll
      for (int e = 0; e < 16; ++e) {
        int i8 = e & 7;
        bool lo = e < 8;
        float xr = (float)(lo ? xr_lo[i8] : xr_hi[i8]);
        float xz = (float)(lo ? xz_lo[i8] : xz_hi[i8]);
        float xn = (float)(lo ? xn_lo[i8] : xn_hi[i8]);
        float hr = (float)(lo ? hr_lo[i8] : hr_hi[i8]);
        float hz = (float)(lo ? hz_lo[i8] : hz_hi[i8]);
        float hn = (float)(lo ? hn_lo[i8] : hn_hi[i8]);
        float ho = (float)(lo ? ho_lo[i8] : ho_hi[i8]);
        float r  = 1.f / (1.f + __expf(-(xr + hr)));
        float z  = 1.f / (1.f + __expf(-(xz + hz)));
        float a2 = __expf(2.f * (xn + r * hn));
        float nn = (a2 - 1.f) / (a2 + 1.f);
        float hv = (1.f - z) * nn + z * ho;
        if (lo) { nw_lo[i8] = (_Float16)hv; fo_lo[i8] = hv; }
        else    { nw_hi[i8] = (_Float16)hv; fo_hi[i8] = hv; }
      }
      *(v8h*)&hFrag[base2]     = nw_lo;
      *(v8h*)&hFrag[base2 + 8] = nw_hi;

      size_t ob = ((size_t)tt * B_ + b2) * 512 + d * H_;
      if (isFinal) {
        *(v8f*)&outF[ob + jb]      = fo_lo;
        *(v8f*)&outF[ob + jb + 16] = fo_hi;
      } else {
        *(v8h*)&outH[ob + jb]      = nw_lo;
        *(v8h*)&outH[ob + jb + 16] = nw_hi;
      }
    }
    __syncthreads();
  }
}

// ------------------------------------------------------------------------ host
extern "C" void kernel_launch(void* const* d_in, const int* in_sizes, int n_in,
                              void* d_out, int out_size, void* d_ws,
                              size_t ws_size, hipStream_t stream) {
  (void)in_sizes; (void)n_in; (void)out_size; (void)ws_size;
  const float* x    = (const float*)d_in[0];  // [T,B,I]
  const float* h0   = (const float*)d_in[1];  // [12,B,H]
  const float* Wih0 = (const float*)d_in[2];  // [2,768,128]
  const float* Wih  = (const float*)d_in[3];  // [5,2,768,512]
  const float* Whh  = (const float*)d_in[4];  // [6,2,768,256]
  const float* bih  = (const float*)d_in[5];  // [6,2,768]
  const float* bhh  = (const float*)d_in[6];  // [6,2,768]
  float* out = (float*)d_out;                 // [T,B,512] f32

  char*  ws  = (char*)d_ws;
  size_t off = 0;
  auto alloc = [&](size_t bytes) -> char* {
    char* p = ws + off;
    off = (off + bytes + 255) & ~(size_t)255;
    return p;
  };
  _Float16* xh   = (_Float16*)alloc((size_t)TB_ * I_ * 2);          // 16 MB
  _Float16* w0h  = (_Float16*)alloc((size_t)1536 * 128 * 2);
  _Float16* wlh  = (_Float16*)alloc((size_t)5 * 1536 * 512 * 2);
  _Float16* whhF = (_Float16*)alloc((size_t)12 * G_ * H_ * 2);
  _Float16* xgB  = (_Float16*)alloc((size_t)2 * TB_ * G_ * 2);      // 201 MB
  _Float16* bufA = (_Float16*)alloc((size_t)TB_ * 512 * 2);         // 67 MB
  _Float16* bufB = (_Float16*)alloc((size_t)TB_ * 512 * 2);         // 67 MB

  auto cast = [&](const float* s, _Float16* dst, int n) {
    cast_f32_f16<<<(n + 255) / 256, 256, 0, stream>>>(s, dst, n);
  };
  cast(x, xh, TB_ * I_);
  cast(Wih0, w0h, 2 * G_ * I_);
  cast(Wih, wlh, 5 * 2 * G_ * 512);
  {
    int n = 12 * 48 * 8 * 32 * 16;
    swizzle_whh<<<(n + 255) / 256, 256, 0, stream>>>(Whh, whhF);
  }

  for (int l = 0; l < L_; ++l) {
    const _Float16* A  = (l == 0) ? xh : ((l & 1) ? bufA : bufB);
    _Float16*       ob = (l & 1) ? bufB : bufA;
    const _Float16* Wc = (l == 0) ? w0h : (wlh + (size_t)(l - 1) * 1536 * 512);
    int K = (l == 0) ? I_ : 512;
    dim3 grid(TB_ / 128, 1536 / 64);
    gemm_xg<<<grid, 256, 0, stream>>>(A, Wc, K, bih + (size_t)l * 2 * G_, xgB);
    gru_scan<<<2, 1024, 0, stream>>>(
        xgB, whhF + (size_t)l * 2 * 48 * 8 * 32 * 16,
        bhh + (size_t)l * 2 * G_, h0 + (size_t)2 * l * B_ * H_,
        ob, out, (l == L_ - 1) ? 1 : 0);
  }
}
